// FFLayer_14456859918385
// MI455X (gfx1250) — compile-verified, matrix operand order checked by reading, not run on hardware
//
#include <hip/hip_runtime.h>
#include <hip/hip_bf16.h>
#include <stdint.h>

// ---------------------------------------------------------------------------
// Types for WMMA
// ---------------------------------------------------------------------------
typedef __attribute__((ext_vector_type(16))) __bf16 v16bf;
typedef __attribute__((ext_vector_type(8)))  float  v8f;

#define B_ROWS 8192
#define K_DIM  4096
#define N_DIM  4096
#define EPSV   1e-4f

// round-to-nearest-even fp32 -> bf16
static __device__ __forceinline__ unsigned short f2bf(float f) {
    unsigned u = __builtin_bit_cast(unsigned, f);
    u += 0x7FFFu + ((u >> 16) & 1u);
    return (unsigned short)(u >> 16);
}

// ---------------------------------------------------------------------------
// Kernel 1: per-row L2 normalize x, output bf16 row-major [B_ROWS x K_DIM]
// One 256-thread block (8 wave32) per row; each thread handles 16 elements.
// ---------------------------------------------------------------------------
__global__ void __launch_bounds__(256)
ffl_normalize_rows(const float* __restrict__ x, unsigned short* __restrict__ xb) {
    const int row  = blockIdx.x;
    const int tid  = threadIdx.x;
    const float* xr = x + (size_t)row * K_DIM;

    // 16 contiguous floats per thread (4 x float4), fully coalesced per wave
    float4 v0 = ((const float4*)xr)[tid * 4 + 0];
    float4 v1 = ((const float4*)xr)[tid * 4 + 1];
    float4 v2 = ((const float4*)xr)[tid * 4 + 2];
    float4 v3 = ((const float4*)xr)[tid * 4 + 3];

    float s = v0.x*v0.x + v0.y*v0.y + v0.z*v0.z + v0.w*v0.w
            + v1.x*v1.x + v1.y*v1.y + v1.z*v1.z + v1.w*v1.w
            + v2.x*v2.x + v2.y*v2.y + v2.z*v2.z + v2.w*v2.w
            + v3.x*v3.x + v3.y*v3.y + v3.z*v3.z + v3.w*v3.w;

    // wave32 butterfly reduction
    #pragma unroll
    for (int off = 16; off > 0; off >>= 1)
        s += __shfl_xor(s, off, 32);

    __shared__ float wsum[8];
    const int wid  = tid >> 5;
    const int lane = tid & 31;
    if (lane == 0) wsum[wid] = s;
    __syncthreads();

    __shared__ float inv_scale;
    if (tid == 0) {
        float t = wsum[0] + wsum[1] + wsum[2] + wsum[3]
                + wsum[4] + wsum[5] + wsum[6] + wsum[7];
        inv_scale = 1.0f / (__builtin_sqrtf(t) + EPSV);
    }
    __syncthreads();
    const float sc = inv_scale;

    // convert 16 floats -> 16 bf16 -> two uint4 stores
    unsigned short h[16];
    h[ 0]=f2bf(v0.x*sc); h[ 1]=f2bf(v0.y*sc); h[ 2]=f2bf(v0.z*sc); h[ 3]=f2bf(v0.w*sc);
    h[ 4]=f2bf(v1.x*sc); h[ 5]=f2bf(v1.y*sc); h[ 6]=f2bf(v1.z*sc); h[ 7]=f2bf(v1.w*sc);
    h[ 8]=f2bf(v2.x*sc); h[ 9]=f2bf(v2.y*sc); h[10]=f2bf(v2.z*sc); h[11]=f2bf(v2.w*sc);
    h[12]=f2bf(v3.x*sc); h[13]=f2bf(v3.y*sc); h[14]=f2bf(v3.z*sc); h[15]=f2bf(v3.w*sc);

    uint4* dst = (uint4*)(xb + (size_t)row * K_DIM + tid * 16);
    dst[0] = *(const uint4*)&h[0];
    dst[1] = *(const uint4*)&h[8];
}

// ---------------------------------------------------------------------------
// Kernel 2: convert W fp32 -> bf16, 8 elements per thread
// ---------------------------------------------------------------------------
__global__ void __launch_bounds__(256)
ffl_convert_w(const float* __restrict__ w, unsigned short* __restrict__ wb) {
    const size_t base = ((size_t)blockIdx.x * 256 + threadIdx.x) * 8;
    float4 a = *(const float4*)(w + base);
    float4 b = *(const float4*)(w + base + 4);
    unsigned short h[8];
    h[0]=f2bf(a.x); h[1]=f2bf(a.y); h[2]=f2bf(a.z); h[3]=f2bf(a.w);
    h[4]=f2bf(b.x); h[5]=f2bf(b.y); h[6]=f2bf(b.z); h[7]=f2bf(b.w);
    *(uint4*)(wb + base) = *(const uint4*)&h[0];
}

// ---------------------------------------------------------------------------
// Kernel 3: WMMA GEMM  out[M,N] = relu(Xb[M,K] * Wb[N,K]^T + bias)
// Block tile 128x256, 256 threads = 8 wave32 arranged 2(M) x 4(N).
// Each wave: 64x64 tile = 4x4 WMMA 16x16 accumulators, K step = 32 (bf16).
// Per K-step: 16 b128 loads feed 16 WMMAs (1.0 vmem per wmma).
//
// Addressing: uniform base pointer (SGPR pair) + per-lane 32-bit byte
// offsets (xb/wb each < 4 GB) so the backend uses the saddr+voffset global
// load form with cheap 32-bit offset increments instead of 64-bit vaddr
// pointer pairs.
//
// A-fragment (16-bit, 16x32) per lane L:
//   row M = L%16 ; two 8-element K chunks at k0 = (L>=16 ? 8 : 0) and k0+16.
// B-fragment (32x16) mirrors this with N = L%16 (W is K-major per row, same
// load pattern). Each chunk = one 128-bit load.
// C layout: VGPR j -> M = j + 8*(L>=16), N = L%16.
// ---------------------------------------------------------------------------
__global__ void __launch_bounds__(256)
ffl_wmma_gemm(const unsigned short* __restrict__ xb,
              const unsigned short* __restrict__ wb,
              const float* __restrict__ bias,
              float* __restrict__ out) {
    const int tid   = threadIdx.x;
    const int lane  = tid & 31;
    const int wid   = tid >> 5;
    const int waveM = wid >> 2;           // 0..1
    const int waveN = wid & 3;            // 0..3

    const int m0 = blockIdx.y * 128 + waveM * 64;   // wave M base (4 subtiles)
    const int n0 = blockIdx.x * 256 + waveN * 64;   // wave N base (4 subtiles)

    const int r  = lane & 15;             // row/col within 16
    const int hi = lane >> 4;             // K-chunk selector

    const char* xbase = (const char*)xb;
    const char* wbase = (const char*)wb;

    // Per-lane 32-bit byte offsets for A (4 M-subtiles) and B (4 N-subtiles).
    unsigned aoff[4];
    #pragma unroll
    for (int mt = 0; mt < 4; ++mt)
        aoff[mt] = (unsigned)(((m0 + mt * 16 + r) * K_DIM + hi * 8) * 2);
    unsigned boff[4];
    #pragma unroll
    for (int nt = 0; nt < 4; ++nt)
        boff[nt] = (unsigned)(((n0 + nt * 16 + r) * K_DIM + hi * 8) * 2);

    // bias per N-subtile (issued early, consumed in epilogue)
    float bn[4];
    #pragma unroll
    for (int nt = 0; nt < 4; ++nt)
        bn[nt] = bias[n0 + nt * 16 + r];

    v8f acc[4][4];
    #pragma unroll
    for (int mt = 0; mt < 4; ++mt)
        #pragma unroll
        for (int nt = 0; nt < 4; ++nt)
            acc[mt][nt] = (v8f){0.f,0.f,0.f,0.f,0.f,0.f,0.f,0.f};

    // K loop: 64 bytes of K per step (32 bf16)
    for (unsigned kbB = 0; kbB < K_DIM * 2; kbB += 64) {
        v16bf afrag[4], bfrag[4];
        // Load A fragments: two b128 chunks -> 16 bf16 per lane
        #pragma unroll
        for (int mt = 0; mt < 4; ++mt) {
            uint4* f = (uint4*)&afrag[mt];
            f[0] = *(const uint4*)(xbase + aoff[mt] + kbB);
            f[1] = *(const uint4*)(xbase + aoff[mt] + kbB + 32);
        }
        #pragma unroll
        for (int nt = 0; nt < 4; ++nt) {
            uint4* f = (uint4*)&bfrag[nt];
            f[0] = *(const uint4*)(wbase + boff[nt] + kbB);
            f[1] = *(const uint4*)(wbase + boff[nt] + kbB + 32);
        }
        // 16 WMMAs: 64x64 tile advance by K=32
        #pragma unroll
        for (int mt = 0; mt < 4; ++mt)
            #pragma unroll
            for (int nt = 0; nt < 4; ++nt)
                acc[mt][nt] = __builtin_amdgcn_wmma_f32_16x16x32_bf16(
                    /*neg_a=*/false, afrag[mt],
                    /*neg_b=*/false, bfrag[nt],
                    /*c_mod=*/(short)0, acc[mt][nt],
                    /*reuse_a=*/false, /*reuse_b=*/false);
    }

    // Epilogue: bias + ReLU + store fp32
    #pragma unroll
    for (int mt = 0; mt < 4; ++mt) {
        #pragma unroll
        for (int nt = 0; nt < 4; ++nt) {
            const int col = n0 + nt * 16 + r;
            #pragma unroll
            for (int j = 0; j < 8; ++j) {
                const int row = m0 + mt * 16 + j + hi * 8;
                float v = acc[mt][nt][j] + bn[nt];
                out[(size_t)row * N_DIM + col] = v > 0.f ? v : 0.f;
            }
        }
    }
}

// ---------------------------------------------------------------------------
// Launcher
// ---------------------------------------------------------------------------
extern "C" void kernel_launch(void* const* d_in, const int* in_sizes, int n_in,
                              void* d_out, int out_size, void* d_ws, size_t ws_size,
                              hipStream_t stream) {
    const float* x = (const float*)d_in[0];   // [8192, 4096]
    const float* W = (const float*)d_in[1];   // [4096, 4096]
    const float* b = (const float*)d_in[2];   // [4096]
    float* out = (float*)d_out;               // [8192, 4096]

    unsigned short* xb = (unsigned short*)d_ws;                       // 64 MB
    unsigned short* wb = xb + (size_t)B_ROWS * K_DIM;                 // +32 MB

    // 1) normalize rows -> bf16
    ffl_normalize_rows<<<B_ROWS, 256, 0, stream>>>(x, xb);

    // 2) W -> bf16
    ffl_convert_w<<<(N_DIM * K_DIM) / (256 * 8), 256, 0, stream>>>(W, wb);

    // 3) WMMA GEMM + bias + ReLU
    dim3 grid(N_DIM / 256, B_ROWS / 128);   // (16, 64)
    ffl_wmma_gemm<<<grid, 256, 0, stream>>>(xb, wb, b, out);
}
